// trajectory2seq_18038862644006
// MI455X (gfx1250) — matrix operand order repaired and checked by reading, not compiled, not run on hardware
//
#include <hip/hip_runtime.h>
#include <hip/hip_bf16.h>
#include <math.h>

typedef unsigned short u16;
typedef __attribute__((ext_vector_type(16))) __bf16 v16bf;
typedef __attribute__((ext_vector_type(8)))  __bf16 v8bf;
typedef __attribute__((ext_vector_type(8)))  float  v8f;

#define HID  512
#define VOC  32000
#define BAT  64
#define SEQL 256
#define DECL 32
#define NCHUNK 125   // 32000 / (8 waves * 2 tiles * 16 cols)

// ---------- helpers ----------

__device__ __forceinline__ u16 f2bf(float f) {
    unsigned u = __float_as_uint(f);
    u += 0x7FFFu + ((u >> 16) & 1u);     // round-to-nearest-even
    return (u16)(u >> 16);
}

// A fragment: 16x32 bf16, row-major source. lane&15 = M row,
// kh = lane>>4 selects K-halves (contiguous, matching B):
//   lanes 0-15 : elems 0..15 -> K 0..15
//   lanes 16-31: elems 0..15 -> K 16..31
__device__ __forceinline__ v16bf load_a_frag(const u16* __restrict__ row, int k0, int kh) {
    v8bf lo = *(const v8bf*)(row + k0 + kh * 16);
    v8bf hi = *(const v8bf*)(row + k0 + kh * 16 + 8);
    v16bf a;
#pragma unroll
    for (int i = 0; i < 8; i++) { a[i] = lo[i]; a[i + 8] = hi[i]; }
    return a;
}

// B fragment: 32x16 bf16 = W^T where W is row-major [N][K].
// lane&15 = column N, lane>>4 = K group; 16 contiguous bf16 per lane.
__device__ __forceinline__ v16bf load_b_frag(const u16* __restrict__ wrow, int k0, int g) {
    return *(const v16bf*)(wrow + k0 + g * 16);
}

#define WMMA_BF16(a, b, c) \
    __builtin_amdgcn_wmma_f32_16x16x32_bf16(false, (a), false, (b), (short)0, (c), false, false)

// K=512 GEMM, one tile, two independent accumulator chains (depth 8 each).
__device__ __forceinline__ v8f gemm512_dual(const u16* __restrict__ aRow,
                                            const u16* __restrict__ wRow, int kh) {
    v8f acc0 = {}, acc1 = {};
#pragma unroll
    for (int k0 = 0; k0 < HID; k0 += 64) {
        acc0 = WMMA_BF16(load_a_frag(aRow, k0, kh),      load_b_frag(wRow, k0, kh),      acc0);
        acc1 = WMMA_BF16(load_a_frag(aRow, k0 + 32, kh), load_b_frag(wRow, k0 + 32, kh), acc1);
    }
    return acc0 + acc1;
}

// K=512 GEMM, TWO output tiles sharing the same A fragments (same M-tile,
// adjacent N-tiles). 4 independent WMMA chains of depth 8; A loaded once.
__device__ __forceinline__ void gemm512_x2(const u16* __restrict__ aRow,
                                           const u16* __restrict__ w0Row,
                                           const u16* __restrict__ w1Row,
                                           int kh, v8f& out0, v8f& out1) {
    v8f t0a = {}, t0b = {}, t1a = {}, t1b = {};
#pragma unroll
    for (int k0 = 0; k0 < HID; k0 += 64) {
        v16bf aA = load_a_frag(aRow, k0, kh);
        v16bf aB = load_a_frag(aRow, k0 + 32, kh);
        t0a = WMMA_BF16(aA, load_b_frag(w0Row, k0, kh),      t0a);
        t1a = WMMA_BF16(aA, load_b_frag(w1Row, k0, kh),      t1a);
        t0b = WMMA_BF16(aB, load_b_frag(w0Row, k0 + 32, kh), t0b);
        t1b = WMMA_BF16(aB, load_b_frag(w1Row, k0 + 32, kh), t1b);
    }
    out0 = t0a + t0b;
    out1 = t1a + t1b;
}

// grid barrier across the 8 persistent encoder workgroups
__device__ __forceinline__ void gbar(unsigned* cnt, unsigned* gen) {
    __syncthreads();
    if (threadIdx.x == 0) {
        __threadfence();
        unsigned g = __hip_atomic_load(gen, __ATOMIC_RELAXED, __HIP_MEMORY_SCOPE_AGENT);
        unsigned a = __hip_atomic_fetch_add(cnt, 1u, __ATOMIC_ACQ_REL, __HIP_MEMORY_SCOPE_AGENT);
        if (a == 7u) {
            __hip_atomic_store(cnt, 0u, __ATOMIC_RELAXED, __HIP_MEMORY_SCOPE_AGENT);
            __hip_atomic_fetch_add(gen, 1u, __ATOMIC_RELEASE, __HIP_MEMORY_SCOPE_AGENT);
        } else {
            while (__hip_atomic_load(gen, __ATOMIC_ACQUIRE, __HIP_MEMORY_SCOPE_AGENT) == g)
                __builtin_amdgcn_s_sleep(1);
        }
    }
    __syncthreads();
}

// ---------- kernels ----------

__global__ void k_cvt(const float* __restrict__ src, u16* __restrict__ dst, size_t n) {
    size_t i = (size_t)blockIdx.x * blockDim.x + threadIdx.x;
    size_t s = (size_t)gridDim.x * blockDim.x;
    for (; i < n; i += s) dst[i] = f2bf(src[i]);
}

// P0[t*B + b][h] = emb[x[b,t]] @ W_ih0^T   (bias added later)
// pair of adjacent N-tiles per wave, shared A.
__global__ void __launch_bounds__(256) k_p0(const int* __restrict__ x,
                                            const u16* __restrict__ embb,
                                            const u16* __restrict__ wih0,
                                            float* __restrict__ P0) {
    int lane = threadIdx.x & 31, ln = lane & 15, kh = lane >> 4;
    int pair = blockIdx.x * 8 + (threadIdx.x >> 5);  // 0..16383
    int mt = pair >> 4;                              // 1024 m-tiles
    int nt0 = (pair & 15) * 2;                       // 16 n-tile pairs
    int m0 = mt * 16, n00 = nt0 * 16, n01 = n00 + 16;
    int mrow = m0 + ln;                              // global row = t*64 + b
    int t = mrow >> 6, b = mrow & 63;
    const u16* aRow = embb + (size_t)x[(size_t)b * SEQL + t] * HID;
    const u16* w0 = wih0 + (size_t)(n00 + ln) * HID;
    const u16* w1 = wih0 + (size_t)(n01 + ln) * HID;
    v8f acc0, acc1;
    gemm512_x2(aRow, w0, w1, kh, acc0, acc1);
#pragma unroll
    for (int r = 0; r < 8; r++) {
        int gm = m0 + kh * 8 + r;
        P0[(size_t)gm * HID + n00 + ln] = acc0[r];
        P0[(size_t)gm * HID + n01 + ln] = acc1[r];
    }
}

// persistent 256-step, 2-layer recurrence; 8 WGs * 8 waves;
// each wave: one tile-pair per layer (shared A fragments).
__global__ void __launch_bounds__(256) k_encoder(
        const float* __restrict__ P0,
        const u16* __restrict__ wih1, const u16* __restrict__ whh0,
        const u16* __restrict__ whh1,
        const float* __restrict__ bih, const float* __restrict__ bhh,
        u16* h0a, u16* h0b, u16* h1a, u16* h1b,
        unsigned* cnt, unsigned* gen) {
    int lane = threadIdx.x & 31, ln = lane & 15, kh = lane >> 4;
    int wid = blockIdx.x * 8 + (threadIdx.x >> 5);   // 0..63 -> tile pair (2w,2w+1)
    int mt = (wid * 2) >> 5, nt0 = (wid * 2) & 31;   // pair never crosses m-tile
    int m0 = mt * 16, n00 = nt0 * 16, n01 = n00 + 16;
    for (int t = 0; t < SEQL; t++) {
        const u16* h0s = (t & 1) ? h0b : h0a;
        u16*       h0d = (t & 1) ? h0a : h0b;
        const u16* h1s = (t & 1) ? h1b : h1a;
        u16*       h1d = (t & 1) ? h1a : h1b;
        const float* p0t = P0 + (size_t)t * BAT * HID;
        // layer 0: h0' = tanh(P0 + b_ih0 + h0 @ Whh0^T + b_hh0)
        {
            const u16* aRow = h0s + (size_t)(m0 + ln) * HID;
            v8f acc0, acc1;
            gemm512_x2(aRow, whh0 + (size_t)(n00 + ln) * HID,
                             whh0 + (size_t)(n01 + ln) * HID, kh, acc0, acc1);
#pragma unroll
            for (int r = 0; r < 8; r++) {
                int gm = m0 + kh * 8 + r;
                int g0 = n00 + ln, g1 = n01 + ln;
                float v0 = acc0[r] + p0t[(size_t)gm * HID + g0] + bih[g0] + bhh[g0];
                float v1 = acc1[r] + p0t[(size_t)gm * HID + g1] + bih[g1] + bhh[g1];
                h0d[(size_t)gm * HID + g0] = f2bf(tanhf(v0));
                h0d[(size_t)gm * HID + g1] = f2bf(tanhf(v1));
            }
        }
        gbar(cnt, gen);
        // layer 1: h1' = tanh(h0' @ Wih1^T + b_ih1 + h1 @ Whh1^T + b_hh1)
        {
            const u16* a1 = h0d + (size_t)(m0 + ln) * HID;
            const u16* a2 = h1s + (size_t)(m0 + ln) * HID;
            v8f i0, i1, r0, r1;
            gemm512_x2(a1, wih1 + (size_t)(n00 + ln) * HID,
                           wih1 + (size_t)(n01 + ln) * HID, kh, i0, i1);
            gemm512_x2(a2, whh1 + (size_t)(n00 + ln) * HID,
                           whh1 + (size_t)(n01 + ln) * HID, kh, r0, r1);
#pragma unroll
            for (int r = 0; r < 8; r++) {
                int gm = m0 + kh * 8 + r;
                int g0 = n00 + ln, g1 = n01 + ln;
                float v0 = i0[r] + r0[r] + bih[HID + g0] + bhh[HID + g0];
                float v1 = i1[r] + r1[r] + bih[HID + g1] + bhh[HID + g1];
                h1d[(size_t)gm * HID + g0] = f2bf(tanhf(v0));
                h1d[(size_t)gm * HID + g1] = f2bf(tanhf(v1));
            }
        }
        gbar(cnt, gen);
    }
}

// decoder layer 0: A rows gathered from emb via tok; 1 tile/wave, K-split chains
__global__ void __launch_bounds__(256) k_dec_l0(
        const u16* __restrict__ embb, const int* __restrict__ tok,
        const u16* __restrict__ wih0, const u16* __restrict__ whh0,
        const float* __restrict__ bi, const float* __restrict__ bh,
        const u16* __restrict__ h0s, u16* __restrict__ h0d, float* __restrict__ h0f) {
    int lane = threadIdx.x & 31, ln = lane & 15, kh = lane >> 4;
    int wid = blockIdx.x * 8 + (threadIdx.x >> 5);   // 0..127 -> 4x32 tiles
    int mt = wid >> 5, nt = wid & 31;
    int m0 = mt * 16, n0 = nt * 16;
    const u16* aE = embb + (size_t)tok[m0 + ln] * HID;
    const u16* aH = h0s + (size_t)(m0 + ln) * HID;
    v8f acc1 = gemm512_dual(aE, wih0 + (size_t)(n0 + ln) * HID, kh);
    v8f acc2 = gemm512_dual(aH, whh0 + (size_t)(n0 + ln) * HID, kh);
#pragma unroll
    for (int r = 0; r < 8; r++) {
        int gm = m0 + kh * 8 + r, gn = n0 + ln;
        float v = tanhf(acc1[r] + acc2[r] + bi[gn] + bh[gn]);
        h0d[(size_t)gm * HID + gn] = f2bf(v);
        h0f[(size_t)gm * HID + gn] = v;
    }
}

// decoder layer 1
__global__ void __launch_bounds__(256) k_dec_l1(
        const u16* __restrict__ h0n, const u16* __restrict__ wih1,
        const u16* __restrict__ h1s, const u16* __restrict__ whh1,
        const float* __restrict__ bi, const float* __restrict__ bh,
        u16* __restrict__ h1d, float* __restrict__ h1f) {
    int lane = threadIdx.x & 31, ln = lane & 15, kh = lane >> 4;
    int wid = blockIdx.x * 8 + (threadIdx.x >> 5);
    int mt = wid >> 5, nt = wid & 31;
    int m0 = mt * 16, n0 = nt * 16;
    const u16* a1 = h0n + (size_t)(m0 + ln) * HID;
    const u16* a2 = h1s + (size_t)(m0 + ln) * HID;
    v8f acc1 = gemm512_dual(a1, wih1 + (size_t)(n0 + ln) * HID, kh);
    v8f acc2 = gemm512_dual(a2, whh1 + (size_t)(n0 + ln) * HID, kh);
#pragma unroll
    for (int r = 0; r < 8; r++) {
        int gm = m0 + kh * 8 + r, gn = n0 + ln;
        float v = tanhf(acc1[r] + acc2[r] + bi[gn] + bh[gn]);
        h1d[(size_t)gm * HID + gn] = f2bf(v);
        h1f[(size_t)gm * HID + gn] = v;
    }
}

// logits GEMM 64x32000 (K=512): tile-pair per wave (256 vocab cols per WG),
// writes straight into d_out + per-block argmax partials.
__global__ void __launch_bounds__(256) k_logits(
        const u16* __restrict__ h1, const u16* __restrict__ wfc,
        const float* __restrict__ bfc, float* __restrict__ out, int step,
        float* __restrict__ pv, int* __restrict__ pi) {
    __shared__ float sv[8][16];
    __shared__ int   si[8][16];
    int lane = threadIdx.x & 31, ln = lane & 15, kh = lane >> 4;
    int w = threadIdx.x >> 5;
    int nb = blockIdx.x;                 // 0..124 (256 vocab cols each)
    int mt = blockIdx.y;                 // 0..3
    int m0 = mt * 16;
    int n00 = (nb * 16 + w * 2) * 16;
    int n01 = n00 + 16;
    const u16* aRow = h1 + (size_t)(m0 + ln) * HID;
    v8f acc0, acc1;
    gemm512_x2(aRow, wfc + (size_t)(n00 + ln) * HID,
                     wfc + (size_t)(n01 + ln) * HID, kh, acc0, acc1);
    float bias0 = bfc[n00 + ln];
    float bias1 = bfc[n01 + ln];
#pragma unroll
    for (int r = 0; r < 8; r++) {
        int gm = m0 + kh * 8 + r;
        int g0 = n00 + ln, g1 = n01 + ln;
        float v0 = acc0[r] + bias0;
        float v1 = acc1[r] + bias1;
        size_t rowoff = (size_t)gm * ((size_t)DECL * VOC) + (size_t)step * VOC;
        out[rowoff + g0] = v0;
        out[rowoff + g1] = v1;
        // best of this lane's two columns (g0 < g1: strictly-greater keeps lower idx)
        float bv = v0; int bi = g0;
        if (v1 > bv) { bv = v1; bi = g1; }
        // reduce max over the 16 lanes of this half (ties -> lowest index)
#pragma unroll
        for (int s = 1; s < 16; s <<= 1) {
            float ov = __shfl_xor(bv, s, 32);
            int   oi = __shfl_xor(bi, s, 32);
            if (ov > bv || (ov == bv && oi < bi)) { bv = ov; bi = oi; }
        }
        if (ln == 0) { sv[w][kh * 8 + r] = bv; si[w][kh * 8 + r] = bi; }
    }
    __syncthreads();
    if (threadIdx.x < 16) {
        int row = threadIdx.x;
        float bv = sv[0][row]; int bi = si[0][row];
#pragma unroll
        for (int ww = 1; ww < 8; ww++) {
            float ov = sv[ww][row]; int oi = si[ww][row];
            if (ov > bv || (ov == bv && oi < bi)) { bv = ov; bi = oi; }
        }
        pv[(size_t)(m0 + row) * NCHUNK + nb] = bv;
        pi[(size_t)(m0 + row) * NCHUNK + nb] = bi;
    }
}

__global__ void k_argmax(const float* __restrict__ pv, const int* __restrict__ pi,
                         int* __restrict__ tok) {
    int b = threadIdx.x;
    if (b >= BAT) return;
    float bv = -INFINITY; int bi = 0;
    for (int c = 0; c < NCHUNK; c++) {
        float v = pv[(size_t)b * NCHUNK + c];
        int   i = pi[(size_t)b * NCHUNK + c];
        if (v > bv || (v == bv && i < bi)) { bv = v; bi = i; }
    }
    tok[b] = bi;
}

__global__ void k_hfin(const float* __restrict__ h0f, const float* __restrict__ h1f,
                       float* __restrict__ out) {
    int i = blockIdx.x * blockDim.x + threadIdx.x;   // 0..65535
    if (i < BAT * HID) out[i] = h0f[i];
    else               out[i] = h1f[i - BAT * HID];
}

// ---------- host ----------

extern "C" void kernel_launch(void* const* d_in, const int* in_sizes, int n_in,
                              void* d_out, int out_size, void* d_ws, size_t ws_size,
                              hipStream_t stream) {
    (void)in_sizes; (void)n_in; (void)out_size; (void)ws_size;
    const int*   x   = (const int*)  d_in[0];
    const float* emb = (const float*)d_in[1];
    const float* Wih = (const float*)d_in[2];
    const float* Whh = (const float*)d_in[3];
    const float* bih = (const float*)d_in[4];
    const float* bhh = (const float*)d_in[5];
    const float* Wfc = (const float*)d_in[6];
    const float* bfc = (const float*)d_in[7];
    float* out = (float*)d_out;

    char* ws = (char*)d_ws;
    size_t off = 0;
    auto take = [&](size_t bytes) -> char* {
        char* p = ws + off; off = (off + bytes + 255) & ~(size_t)255; return p;
    };
    unsigned* bar  = (unsigned*)take(2 * sizeof(unsigned));
    u16* embb = (u16*)take((size_t)VOC * HID * 2);
    u16* wihb = (u16*)take((size_t)2 * HID * HID * 2);
    u16* whhb = (u16*)take((size_t)2 * HID * HID * 2);
    u16* wfcb = (u16*)take((size_t)VOC * HID * 2);
    float* P0 = (float*)take((size_t)SEQL * BAT * HID * 4);
    u16* h0[2] = { (u16*)take((size_t)BAT * HID * 2), (u16*)take((size_t)BAT * HID * 2) };
    u16* h1[2] = { (u16*)take((size_t)BAT * HID * 2), (u16*)take((size_t)BAT * HID * 2) };
    float* h0f = (float*)take((size_t)BAT * HID * 4);
    float* h1f = (float*)take((size_t)BAT * HID * 4);
    int*  tok = (int*)take((size_t)BAT * 4);
    float* pv = (float*)take((size_t)BAT * NCHUNK * 4);
    int*   pi = (int*)take((size_t)BAT * NCHUNK * 4);

    (void)hipMemsetAsync(bar, 0, 2 * sizeof(unsigned), stream);
    (void)hipMemsetAsync(h0[0], 0, (size_t)BAT * HID * 2, stream);
    (void)hipMemsetAsync(h1[0], 0, (size_t)BAT * HID * 2, stream);
    (void)hipMemsetAsync(tok, 0, (size_t)BAT * 4, stream);

    k_cvt<<<2048, 256, 0, stream>>>(emb, embb, (size_t)VOC * HID);
    k_cvt<<<256, 256, 0, stream>>>(Wih, wihb, (size_t)2 * HID * HID);
    k_cvt<<<256, 256, 0, stream>>>(Whh, whhb, (size_t)2 * HID * HID);
    k_cvt<<<2048, 256, 0, stream>>>(Wfc, wfcb, (size_t)VOC * HID);

    k_p0<<<2048, 256, 0, stream>>>(x, embb, wihb /*layer0*/, P0);
    k_encoder<<<8, 256, 0, stream>>>(P0, wihb + (size_t)HID * HID, whhb,
                                     whhb + (size_t)HID * HID, bih, bhh,
                                     h0[0], h0[1], h1[0], h1[1], bar, bar + 1);

    for (int d = 0; d < DECL; d++) {
        int p = d & 1;
        k_dec_l0<<<16, 256, 0, stream>>>(embb, tok, wihb, whhb, bih, bhh,
                                         h0[p], h0[1 - p], h0f);
        k_dec_l1<<<16, 256, 0, stream>>>(h0[1 - p], wihb + (size_t)HID * HID,
                                         h1[p], whhb + (size_t)HID * HID,
                                         bih + HID, bhh + HID, h1[1 - p], h1f);
        k_logits<<<dim3(NCHUNK, 4), 256, 0, stream>>>(h1[1 - p], wfcb, bfc, out, d, pv, pi);
        k_argmax<<<1, 64, 0, stream>>>(pv, pi, tok);
    }
    k_hfin<<<(2 * BAT * HID) / 256, 256, 0, stream>>>(h0f, h1f,
                                                      out + (size_t)BAT * DECL * VOC);
}
